// MoE_25099788878261
// MI455X (gfx1250) — matrix operand order repaired and checked by reading, not compiled, not run on hardware
//
#include <hip/hip_runtime.h>

typedef __bf16 bf16_t;
typedef __attribute__((ext_vector_type(16))) __bf16 v16bf;
typedef __attribute__((ext_vector_type(8)))  __bf16 v8bf;
typedef __attribute__((ext_vector_type(8)))  float  v8f;

#define HIDDEN 1024
#define NEXP   8
#define INTER  1408
#define SI     2816
#define T_TOK  4096
#define TM     32     // token rows per block (2 WMMA M-tiles)

union V16 { v16bf v; v8bf h[2]; };

// A-fragment (16x32 bf16, row-major source): lane m = lane&15 picks the row,
// lane>=16 shifts K by 8; elements 0..7 = K kb..kb+7, elements 8..15 = K kb+16..kb+23.
__device__ __forceinline__ v16bf load_frag_a(const bf16_t* rowptr, int k0, int lane) {
  int kb = (lane >> 4) * 8;
  V16 r;
  r.h[0] = *(const v8bf*)(rowptr + k0 + kb);
  r.h[1] = *(const v8bf*)(rowptr + k0 + kb + 16);
  return r.v;
}
// B-fragment (32x16 bf16) from K-contiguous (pre-transposed) weights:
// lane>=16 shifts K by 8; elements 0..7 = K kb..kb+7, elements 8..15 = K kb+16..kb+23.
__device__ __forceinline__ v16bf load_frag_b(const bf16_t* colptr, int k0, int lane) {
  int kb = (lane >> 4) * 8;
  V16 r;
  r.h[0] = *(const v8bf*)(colptr + k0 + kb);
  r.h[1] = *(const v8bf*)(colptr + k0 + kb + 16);
  return r.v;
}

__device__ __forceinline__ v8f wmma_bf16(v16bf a, v16bf b, v8f c) {
  return __builtin_amdgcn_wmma_f32_16x16x32_bf16(false, a, false, b, (short)0, c, false, false);
}

// ---------------- routing (fp32, matches reference MoEGate) ----------------
__global__ void zero_cnt_kernel(int* cnt) {
  if (threadIdx.x < NEXP) cnt[threadIdx.x] = 0;
}

__global__ __launch_bounds__(256) void gate_kernel(
    const float* __restrict__ x, const float* __restrict__ gw,
    int* __restrict__ cnt, int* __restrict__ btok, float* __restrict__ bwt) {
  __shared__ float sgw[NEXP * HIDDEN];   // 32 KB
  for (int i = threadIdx.x; i < NEXP * HIDDEN; i += 256) sgw[i] = gw[i];
  __syncthreads();
  int t = blockIdx.x * 256 + threadIdx.x;           // grid = 16 blocks exactly
  const float* xr = x + (size_t)t * HIDDEN;
  float acc[NEXP];
  #pragma unroll
  for (int e = 0; e < NEXP; ++e) acc[e] = 0.f;
  for (int h = 0; h < HIDDEN; ++h) {
    float xv = xr[h];
    #pragma unroll
    for (int e = 0; e < NEXP; ++e) acc[e] += xv * sgw[e * HIDDEN + h];
  }
  float mx = acc[0];
  #pragma unroll
  for (int e = 1; e < NEXP; ++e) mx = fmaxf(mx, acc[e]);
  float sum = 0.f;
  #pragma unroll
  for (int e = 0; e < NEXP; ++e) { acc[e] = __expf(acc[e] - mx); sum += acc[e]; }
  float inv = 1.f / sum;
  #pragma unroll
  for (int e = 0; e < NEXP; ++e) acc[e] *= inv;
  int i0 = 0;
  #pragma unroll
  for (int e = 1; e < NEXP; ++e) if (acc[e] > acc[i0]) i0 = e;
  int i1 = (i0 == 0) ? 1 : 0;
  #pragma unroll
  for (int e = 0; e < NEXP; ++e) if (e != i0 && acc[e] > acc[i1]) i1 = e;
  float w0 = acc[i0], w1 = acc[i1];
  float d = w0 + w1 + 1e-20f;
  w0 /= d; w1 /= d;                                  // SCALE = 1.0
  int p0 = atomicAdd(&cnt[i0], 1);
  btok[i0 * T_TOK + p0] = t * 2 + 0; bwt[i0 * T_TOK + p0] = w0;
  int p1 = atomicAdd(&cnt[i1], 1);
  btok[i1 * T_TOK + p1] = t * 2 + 1; bwt[i1 * T_TOK + p1] = w1;
}

// ---------------- conversions ----------------
__global__ void cvt_x_kernel(const float* __restrict__ x, bf16_t* __restrict__ xb, int n) {
  int i = blockIdx.x * 256 + threadIdx.x;
  if (i < n) xb[i] = (bf16_t)x[i];
}

// in: [R][C] fp32 row-major (per blockIdx.z matrix); out: [C][R] bf16 row-major
__global__ void transpose_cvt_kernel(const float* __restrict__ in, bf16_t* __restrict__ out,
                                     int R, int C) {
  __shared__ float tile[32][33];
  const size_t mat = (size_t)blockIdx.z * (size_t)R * (size_t)C;
  int c0 = blockIdx.x * 32, r0 = blockIdx.y * 32;
  int tx = threadIdx.x, ty = threadIdx.y;
  for (int i = ty; i < 32; i += 8) {
    int r = r0 + i, c = c0 + tx;
    if (r < R && c < C) tile[i][tx] = in[mat + (size_t)r * C + c];
  }
  __syncthreads();
  for (int i = ty; i < 32; i += 8) {
    int c = c0 + i, r = r0 + tx;
    if (r < R && c < C) out[mat + (size_t)c * R + r] = (bf16_t)tile[tx][i];
  }
}

// ---------------- fused SwiGLU, routed experts ----------------
// grid = (T_TOK/TM = 128, NEXP); block = 256 (8 wave32)
__global__ __launch_bounds__(256) void routed_kernel(
    const bf16_t* __restrict__ xb,
    const bf16_t* __restrict__ wgT, const bf16_t* __restrict__ wuT,
    const bf16_t* __restrict__ wdT,
    const int* __restrict__ cnt, const int* __restrict__ btok,
    const float* __restrict__ bwt, float* __restrict__ part) {
  __shared__ bf16_t act[TM * INTER];   // 88 KB (gfx1250 WGP has 320 KB LDS)
  __shared__ int   rOut[TM];
  __shared__ int   rTok[TM];
  __shared__ float rW[TM];
  __shared__ int   rValid[TM];

  const int e = blockIdx.y;
  const int nrows = cnt[e];
  const int tile = blockIdx.x;
  if (tile * TM >= nrows) return;

  const int tid = threadIdx.x;
  if (tid < TM) {
    int i = tile * TM + tid;
    int valid = (i < nrows) ? 1 : 0;
    int ent = valid ? btok[e * T_TOK + i] : 0;
    rValid[tid] = valid;
    rTok[tid] = ent >> 1;
    rOut[tid] = ent;                       // part row = tok*2 + slot
    rW[tid] = valid ? bwt[e * T_TOK + i] : 0.f;
  }
  __syncthreads();

  const int lane = tid & 31;
  const int wave = tid >> 5;
  const int lm = lane & 15;
  const int mhi = (lane >> 4) * 8;

  const bf16_t* b1g = wgT + (size_t)e * INTER * HIDDEN;  // [INTER][HIDDEN]
  const bf16_t* b1u = wuT + (size_t)e * INTER * HIDDEN;
  const bf16_t* b2  = wdT + (size_t)e * HIDDEN * INTER;  // [HIDDEN][INTER]
  const bf16_t* arow0 = xb + (size_t)rTok[lm] * HIDDEN;
  const bf16_t* arow1 = xb + (size_t)rTok[16 + lm] * HIDDEN;

  // phase 1: act = silu(x@Wg) * (x@Wu), 2 M-tiles share each B fragment
  for (int it = 0; it < INTER / 128; ++it) {             // 11 chunks of 8 waves * 16 N
    int n0 = (it * 8 + wave) * 16;
    const bf16_t* bg = b1g + (size_t)(n0 + lm) * HIDDEN;
    const bf16_t* bu = b1u + (size_t)(n0 + lm) * HIDDEN;
    v8f cg0 = {}, cu0 = {}, cg1 = {}, cu1 = {};
    for (int k0 = 0; k0 < HIDDEN; k0 += 32) {
      v16bf a0  = load_frag_a(arow0, k0, lane);
      v16bf a1  = load_frag_a(arow1, k0, lane);
      v16bf vbg = load_frag_b(bg, k0, lane);
      v16bf vbu = load_frag_b(bu, k0, lane);
      cg0 = wmma_bf16(a0, vbg, cg0);
      cg1 = wmma_bf16(a1, vbg, cg1);
      cu0 = wmma_bf16(a0, vbu, cu0);
      cu1 = wmma_bf16(a1, vbu, cu1);
    }
    #pragma unroll
    for (int j = 0; j < 8; ++j) {
      float g0 = cg0[j], u0 = cu0[j];
      float g1 = cg1[j], u1 = cu1[j];
      act[(j + mhi) * INTER + (n0 + lm)]      = (bf16_t)((g0 / (1.f + __expf(-g0))) * u0);
      act[(j + mhi + 16) * INTER + (n0 + lm)] = (bf16_t)((g1 / (1.f + __expf(-g1))) * u1);
    }
  }
  __syncthreads();

  // phase 2: part rows = (act @ Wd) * weight, 2 M-tiles share each B fragment
  const bf16_t* aLds0 = &act[lm * INTER];
  const bf16_t* aLds1 = &act[(16 + lm) * INTER];
  for (int nt = 0; nt < 8; ++nt) {
    int n0 = (wave * 8 + nt) * 16;
    const bf16_t* bcol = b2 + (size_t)(n0 + lm) * INTER;
    v8f c0 = {}, c1 = {};
    for (int k0 = 0; k0 < INTER; k0 += 32) {
      v16bf a0 = load_frag_a(aLds0, k0, lane);
      v16bf a1 = load_frag_a(aLds1, k0, lane);
      v16bf b  = load_frag_b(bcol, k0, lane);
      c0 = wmma_bf16(a0, b, c0);
      c1 = wmma_bf16(a1, b, c1);
    }
    #pragma unroll
    for (int j = 0; j < 8; ++j) {
      int m0 = j + mhi, m1 = j + mhi + 16;
      if (rValid[m0]) part[(size_t)rOut[m0] * HIDDEN + (n0 + lm)] = c0[j] * rW[m0];
      if (rValid[m1]) part[(size_t)rOut[m1] * HIDDEN + (n0 + lm)] = c1[j] * rW[m1];
    }
  }
}

// ---------------- fused SwiGLU, shared expert (SI split into 2 halves) ----------------
// grid = (T_TOK/TM = 128, 2)
__global__ __launch_bounds__(256) void shared_kernel(
    const bf16_t* __restrict__ xb,
    const bf16_t* __restrict__ swgT, const bf16_t* __restrict__ swuT,
    const bf16_t* __restrict__ swdT, float* __restrict__ spart) {
  __shared__ bf16_t act[TM * INTER];   // 88 KB

  const int half = blockIdx.y;
  const int t0 = blockIdx.x * TM;
  const int tid = threadIdx.x;
  const int lane = tid & 31;
  const int wave = tid >> 5;
  const int lm = lane & 15;
  const int mhi = (lane >> 4) * 8;

  const bf16_t* b1g = swgT + (size_t)half * INTER * HIDDEN;  // rows of this half of [SI][HIDDEN]
  const bf16_t* b1u = swuT + (size_t)half * INTER * HIDDEN;
  const bf16_t* b2  = swdT;                                  // [HIDDEN][SI]
  const int k2off = half * INTER;
  const bf16_t* arow0 = xb + (size_t)(t0 + lm) * HIDDEN;
  const bf16_t* arow1 = xb + (size_t)(t0 + 16 + lm) * HIDDEN;

  for (int it = 0; it < INTER / 128; ++it) {
    int n0 = (it * 8 + wave) * 16;
    const bf16_t* bg = b1g + (size_t)(n0 + lm) * HIDDEN;
    const bf16_t* bu = b1u + (size_t)(n0 + lm) * HIDDEN;
    v8f cg0 = {}, cu0 = {}, cg1 = {}, cu1 = {};
    for (int k0 = 0; k0 < HIDDEN; k0 += 32) {
      v16bf a0  = load_frag_a(arow0, k0, lane);
      v16bf a1  = load_frag_a(arow1, k0, lane);
      v16bf vbg = load_frag_b(bg, k0, lane);
      v16bf vbu = load_frag_b(bu, k0, lane);
      cg0 = wmma_bf16(a0, vbg, cg0);
      cg1 = wmma_bf16(a1, vbg, cg1);
      cu0 = wmma_bf16(a0, vbu, cu0);
      cu1 = wmma_bf16(a1, vbu, cu1);
    }
    #pragma unroll
    for (int j = 0; j < 8; ++j) {
      float g0 = cg0[j], u0 = cu0[j];
      float g1 = cg1[j], u1 = cu1[j];
      act[(j + mhi) * INTER + (n0 + lm)]      = (bf16_t)((g0 / (1.f + __expf(-g0))) * u0);
      act[(j + mhi + 16) * INTER + (n0 + lm)] = (bf16_t)((g1 / (1.f + __expf(-g1))) * u1);
    }
  }
  __syncthreads();

  const bf16_t* aLds0 = &act[lm * INTER];
  const bf16_t* aLds1 = &act[(16 + lm) * INTER];
  float* outh = spart + (size_t)half * T_TOK * HIDDEN;
  for (int nt = 0; nt < 8; ++nt) {
    int n0 = (wave * 8 + nt) * 16;
    const bf16_t* bcol = b2 + (size_t)(n0 + lm) * SI + k2off;
    v8f c0 = {}, c1 = {};
    for (int k0 = 0; k0 < INTER; k0 += 32) {
      v16bf a0 = load_frag_a(aLds0, k0, lane);
      v16bf a1 = load_frag_a(aLds1, k0, lane);
      v16bf b  = load_frag_b(bcol, k0, lane);
      c0 = wmma_bf16(a0, b, c0);
      c1 = wmma_bf16(a1, b, c1);
    }
    #pragma unroll
    for (int j = 0; j < 8; ++j) {
      outh[(size_t)(t0 + j + mhi) * HIDDEN + (n0 + lm)]      = c0[j];
      outh[(size_t)(t0 + j + mhi + 16) * HIDDEN + (n0 + lm)] = c1[j];
    }
  }
}

// ---------------- deterministic combine ----------------
__global__ void combine_kernel(const float* __restrict__ part, const float* __restrict__ spart,
                               float* __restrict__ out) {
  size_t t = blockIdx.x;
  for (int h = threadIdx.x; h < HIDDEN; h += 256) {
    out[t * HIDDEN + h] =
        part[(t * 2 + 0) * HIDDEN + h] + part[(t * 2 + 1) * HIDDEN + h] +
        spart[t * HIDDEN + h] + spart[(size_t)T_TOK * HIDDEN + t * HIDDEN + h];
  }
}

extern "C" void kernel_launch(void* const* d_in, const int* in_sizes, int n_in,
                              void* d_out, int out_size, void* d_ws, size_t ws_size,
                              hipStream_t stream) {
  (void)in_sizes; (void)n_in; (void)out_size; (void)ws_size;
  const float* x   = (const float*)d_in[0];
  const float* gw  = (const float*)d_in[1];
  const float* wg  = (const float*)d_in[2];
  const float* wu  = (const float*)d_in[3];
  const float* wd  = (const float*)d_in[4];
  const float* swg = (const float*)d_in[5];
  const float* swu = (const float*)d_in[6];
  const float* swd = (const float*)d_in[7];
  float* out = (float*)d_out;

  char* base = (char*)d_ws;
  size_t off = 0;
  auto alloc = [&](size_t bytes) -> void* {
    void* p = base + off;
    off = (off + bytes + 255) & ~(size_t)255;
    return p;
  };
  int*    cnt   = (int*)   alloc(NEXP * sizeof(int));
  int*    btok  = (int*)   alloc((size_t)NEXP * T_TOK * sizeof(int));
  float*  bwt   = (float*) alloc((size_t)NEXP * T_TOK * sizeof(float));
  bf16_t* xb    = (bf16_t*)alloc((size_t)T_TOK * HIDDEN * 2);
  bf16_t* wgT   = (bf16_t*)alloc((size_t)NEXP * INTER * HIDDEN * 2);
  bf16_t* wuT   = (bf16_t*)alloc((size_t)NEXP * INTER * HIDDEN * 2);
  bf16_t* wdT   = (bf16_t*)alloc((size_t)NEXP * HIDDEN * INTER * 2);
  bf16_t* swgT  = (bf16_t*)alloc((size_t)SI * HIDDEN * 2);
  bf16_t* swuT  = (bf16_t*)alloc((size_t)SI * HIDDEN * 2);
  bf16_t* swdT  = (bf16_t*)alloc((size_t)HIDDEN * SI * 2);
  float*  part  = (float*) alloc((size_t)T_TOK * 2 * HIDDEN * sizeof(float));
  float*  spart = (float*) alloc((size_t)2 * T_TOK * HIDDEN * sizeof(float));

  zero_cnt_kernel<<<1, 32, 0, stream>>>(cnt);
  gate_kernel<<<T_TOK / 256, 256, 0, stream>>>(x, gw, cnt, btok, bwt);
  cvt_x_kernel<<<(T_TOK * HIDDEN + 255) / 256, 256, 0, stream>>>(x, xb, T_TOK * HIDDEN);

  dim3 tb(32, 8);
  transpose_cvt_kernel<<<dim3(INTER / 32, HIDDEN / 32, NEXP), tb, 0, stream>>>(wg, wgT, HIDDEN, INTER);
  transpose_cvt_kernel<<<dim3(INTER / 32, HIDDEN / 32, NEXP), tb, 0, stream>>>(wu, wuT, HIDDEN, INTER);
  transpose_cvt_kernel<<<dim3(HIDDEN / 32, INTER / 32, NEXP), tb, 0, stream>>>(wd, wdT, INTER, HIDDEN);
  transpose_cvt_kernel<<<dim3(SI / 32, HIDDEN / 32, 1), tb, 0, stream>>>(swg, swgT, HIDDEN, SI);
  transpose_cvt_kernel<<<dim3(SI / 32, HIDDEN / 32, 1), tb, 0, stream>>>(swu, swuT, HIDDEN, SI);
  transpose_cvt_kernel<<<dim3(HIDDEN / 32, SI / 32, 1), tb, 0, stream>>>(swd, swdT, SI, HIDDEN);

  routed_kernel<<<dim3(T_TOK / TM, NEXP), 256, 0, stream>>>(xb, wgT, wuT, wdT, cnt, btok, bwt, part);
  shared_kernel<<<dim3(T_TOK / TM, 2), 256, 0, stream>>>(xb, swgT, swuT, swdT, spart);
  combine_kernel<<<T_TOK, 256, 0, stream>>>(part, spart, out);
}
